// HIVAE_Decoder_66073776882429
// MI455X (gfx1250) — matrix operand order repaired and checked by reading, not compiled, not run on hardware
//
#include <hip/hip_runtime.h>
#include <hip/hip_bf16.h>
#include <math.h>

// ---------------------------------------------------------------------------
// HIVAE decoder, MI455X (gfx1250, wave32, WMMA).
// Fused GEMM: C[B x 352] = ys[B x 128] * W[128 x 352]
//   cols   0.. 63 : mean_pz   (Wz^T embedded in rows 64..127, rows 0..63 = 0)
//   cols  64..127 : params_real (f*2 + o)
//   cols 128..351 : cat logits  (f*7 + o)
// GEMM via v_wmma_f32_16x16x32_f16 (f16 in, f32 acc), software-pipelined
// B-fragment loads; epilogue on VALU with prefetched streaming inputs.
// ---------------------------------------------------------------------------

typedef __attribute__((ext_vector_type(16))) _Float16 v16h;
typedef __attribute__((ext_vector_type(8)))  float    v8f;

#define BATCH     131072
#define NTILES    22          // 352 / 16
#define KCHUNKS   4           // 128 / 32
#define FRAG_HALF (NTILES * KCHUNKS * 32 * 16)   // halfs in packed weight blob

#define MEAN_OFF   ((size_t)BATCH * 64)
#define LOGVAR_OFF ((size_t)BATCH * 128)
#define LOGP_OFF   ((size_t)BATCH * 192)

// ---------------------------------------------------------------------------
// Pack combined weight matrix into pre-swizzled wave32 WMMA B-fragments.
// Fragment (tile t, chunk c): lane l covers col n = t*16 + (l&15),
// K = c*32 + (l>>4)*16 + j, j=0..15 contiguous halfs per lane.
// ---------------------------------------------------------------------------
__global__ void hivae_pack_w(const float* __restrict__ Wz,
                             const float* __restrict__ Wreal,
                             const float* __restrict__ Wcat,
                             _Float16* __restrict__ ws)
{
    int tid = blockIdx.x * blockDim.x + threadIdx.x;
    if (tid >= FRAG_HALF) return;
    int j = tid & 15;
    int l = (tid >> 4) & 31;
    int c = (tid >> 9) & 3;
    int t = tid >> 11;
    int n = t * 16 + (l & 15);
    int k = c * 32 + ((l >> 4) << 4) + j;
    float v;
    if (n < 64) {                      // mean_pz: depends on s only
        v = (k < 64) ? 0.0f : Wz[n * 64 + (k - 64)];
    } else if (n < 128) {              // W_real (32,128,2)
        int f = (n - 64) >> 1, o = (n - 64) & 1;
        v = Wreal[(f * 128 + k) * 2 + o];
    } else {                           // W_cat (32,128,7)
        int f = (n - 128) / 7, o = (n - 128) % 7;
        v = Wcat[(f * 128 + k) * 7 + o];
    }
    ws[tid] = (_Float16)v;
}

// ---------------------------------------------------------------------------
// Main kernel: 2 waves / block, each wave owns 16 rows of B.
// ---------------------------------------------------------------------------
__global__ __launch_bounds__(64) void hivae_main(
    const float* __restrict__ y,    const float* __restrict__ s,
    const float* __restrict__ soh,  const float* __restrict__ bz,
    const float* __restrict__ brel, const float* __restrict__ bcat,
    const float* __restrict__ eps,  const float* __restrict__ gum,
    const int*   __restrict__ miss, const _Float16* __restrict__ wfrag,
    float* __restrict__ out)
{
    __shared__ float Csh[2][16][289];   // cols 64..351 staged, pad vs bank conflicts

    const int lane = threadIdx.x & 31;
    const int wv   = threadIdx.x >> 5;
    const int row0 = (blockIdx.x * 2 + wv) * 16;
    const int m    = row0 + (lane & 15);
    const int hi   = lane >> 4;                 // 0 | 1

    // ---- Prefetch this wave's epilogue streaming rows (hide HBM latency
    // behind the GEMM): global_prefetch_b8, speculative, 128B lines.
    {
        const char* psoh = (const char*)(soh + (size_t)row0 * 288);   // 18432 B
        const char* pgum = (const char*)(gum + (size_t)row0 * 256);   // 16384 B
        const char* pmis = (const char*)(miss + (size_t)row0 * 64);   //  4096 B
        const char* peps = (const char*)(eps + (size_t)row0 * 32);    //  2048 B
#pragma unroll
        for (int i = 0; i < 4; ++i) {
            __builtin_prefetch(psoh + (lane + i * 32) * 128, 0, 0);
            __builtin_prefetch(pgum + (lane + i * 32) * 128, 0, 0);
        }
        __builtin_prefetch(pmis + lane * 128, 0, 0);
        __builtin_prefetch(peps + (lane >> 1) * 128, 0, 0);
    }

    // ---- Coalesced zero-fill of log_var_pz for this tile (16 rows x 64)
    {
        float4 z4 = make_float4(0.0f, 0.0f, 0.0f, 0.0f);
        float4* dst = (float4*)(out + LOGVAR_OFF + (size_t)row0 * 64);
#pragma unroll
        for (int i = 0; i < 8; ++i) dst[lane + i * 32] = z4;
    }

    // ---- Build A fragments (contiguous K layout matching B: lane holds
    //      K={b..b+15}, b = hi*16)
    v16h a[4];
#pragma unroll
    for (int c = 0; c < 4; ++c) {
        int kb = c * 32 + hi * 16;
        const float* src = (kb < 64) ? (y + (size_t)m * 64 + kb)
                                     : (s + (size_t)m * 64 + (kb - 64));
        float4 q0 = ((const float4*)src)[0];
        float4 q1 = ((const float4*)src)[1];
        float4 q2 = ((const float4*)(src + 8))[0];
        float4 q3 = ((const float4*)(src + 8))[1];
        v16h av;
        av[0]=(_Float16)q0.x; av[1]=(_Float16)q0.y; av[2]=(_Float16)q0.z; av[3]=(_Float16)q0.w;
        av[4]=(_Float16)q1.x; av[5]=(_Float16)q1.y; av[6]=(_Float16)q1.z; av[7]=(_Float16)q1.w;
        av[8]=(_Float16)q2.x; av[9]=(_Float16)q2.y; av[10]=(_Float16)q2.z; av[11]=(_Float16)q2.w;
        av[12]=(_Float16)q3.x; av[13]=(_Float16)q3.y; av[14]=(_Float16)q3.z; av[15]=(_Float16)q3.w;
        a[c] = av;
    }

    // ---- GEMM over 22 N-tiles, K = 4 x 32, B-fragments double-buffered so
    //      loads for tile t+1 overlap WMMAs of tile t.
    const v16h* bf = (const v16h*)wfrag + lane;

    auto loadB = [&](int t, v16h* b) {
#pragma unroll
        for (int c = 0; c < KCHUNKS; ++c) b[c] = bf[(t * KCHUNKS + c) * 32];
    };
    auto procTile = [&](int t, const v16h* b) {
        v8f acc = {};
#pragma unroll
        for (int c = 0; c < KCHUNKS; ++c)
            acc = __builtin_amdgcn_wmma_f32_16x16x32_f16(
                false, a[c], false, b[c], (short)0, acc, false, false);
        int col = t * 16 + (lane & 15);
        if (t < 4) {
            // mean_pz direct store; C layout: row = v + 8*hi, col = lane%16
            float bzv = bz[col];
            int rb = row0 + hi * 8;
#pragma unroll
            for (int v = 0; v < 8; ++v)
                out[MEAN_OFF + (size_t)(rb + v) * 64 + col] = acc[v] + bzv;
        } else {
            int cl = col - 64;
#pragma unroll
            for (int v = 0; v < 8; ++v)
                Csh[wv][hi * 8 + v][cl] = acc[v];
        }
    };

    v16h bA[KCHUNKS], bB[KCHUNKS];
    loadB(0, bA);
#pragma unroll
    for (int t = 0; t < NTILES; t += 2) {
        loadB(t + 1, bB);
        procTile(t, bA);
        if (t + 2 < NTILES) loadB(t + 2, bA);
        procTile(t + 1, bB);
    }
    __syncthreads();

    // ---- Epilogue: lane f handles real feature f and cat feature f per row
    const float LOG2PI = 1.8378770664093453f;
    const int f = lane;
    const float br0 = brel[2 * f];
    const float br1 = brel[2 * f + 1];
    float bc[7];
#pragma unroll
    for (int j = 0; j < 7; ++j) bc[j] = bcat[7 * f + j];

    for (int r = 0; r < 16; ++r) {
        int rg = row0 + r;
        // -- real feature f
        float mean = Csh[wv][r][2 * f]     + br0;
        float lv   = Csh[wv][r][2 * f + 1] + br1;
        if (lv != lv) lv = 0.0f;
        else if (isinf(lv)) lv = (lv > 0.0f) ? 8.0f : -8.0f;
        lv = fminf(fmaxf(lv, -10.0f), 10.0f);
        float data = soh[(size_t)rg * 288 + f];
        if (data != data) data = 0.0f;
        float mr = (float)miss[(size_t)rg * 64 + f];
        float d  = data - mean;
        float lp = (-0.5f * d * d / expf(lv) - 0.5f * lv - 0.5f * LOG2PI) * mr;
        float st = fmaxf(expf(0.5f * lv), 1e-6f);
        out[(size_t)rg * 64 + f] = mean + st * eps[(size_t)rg * 32 + f];

        // -- cat feature f (K = 8; logit 0 pinned to 0)
        float lpi[8];
        lpi[0] = 0.0f;
#pragma unroll
        for (int j = 1; j < 8; ++j)
            lpi[j] = Csh[wv][r][64 + 7 * f + (j - 1)] + bc[j - 1];
        float mx = lpi[0];
#pragma unroll
        for (int j = 1; j < 8; ++j) mx = fmaxf(mx, lpi[j]);
        float se = 0.0f;
#pragma unroll
        for (int j = 0; j < 8; ++j) se += expf(lpi[j] - mx);
        float lse = mx + logf(se);
        float mc  = (float)miss[(size_t)rg * 64 + 32 + f];
        float lpc = 0.0f;
#pragma unroll
        for (int j = 0; j < 8; ++j)
            lpc += soh[(size_t)rg * 288 + 32 + 8 * f + j] * (lpi[j] - lse);
        lpc *= mc;

        // Gumbel argmax (softmax argmax == perturbed-logit argmax, first-max ties)
        int amax = 0; float best = 0.0f;
#pragma unroll
        for (int j = 0; j < 8; ++j) {
            float u = gum[(size_t)rg * 256 + 8 * f + j];
            u = fminf(fmaxf(u, 1e-6f), 1.0f - 1e-6f);
            float g = -logf(-logf(u));
            float vv = lpi[j] + g;
            if (j == 0 || vv > best) { best = vv; amax = j; }
        }
        out[(size_t)rg * 64 + 32 + f] = (float)amax;

        // -- per-row reduction across 32 lanes -> log_p_x_sum
        float tot = lp + lpc;
#pragma unroll
        for (int off = 16; off > 0; off >>= 1)
            tot += __shfl_xor(tot, off, 32);
        if (lane == 0) out[LOGP_OFF + rg] = tot;
    }
}

extern "C" void kernel_launch(void* const* d_in, const int* in_sizes, int n_in,
                              void* d_out, int out_size, void* d_ws, size_t ws_size,
                              hipStream_t stream) {
    const float* y     = (const float*)d_in[0];
    const float* s     = (const float*)d_in[1];
    const float* soh   = (const float*)d_in[2];
    const float* Wz    = (const float*)d_in[3];
    const float* bz    = (const float*)d_in[4];
    const float* Wreal = (const float*)d_in[5];
    const float* brel  = (const float*)d_in[6];
    const float* Wcat  = (const float*)d_in[7];
    const float* bcat  = (const float*)d_in[8];
    const float* eps   = (const float*)d_in[9];
    const float* gum   = (const float*)d_in[10];
    const int*   miss  = (const int*)d_in[11];
    float* out = (float*)d_out;
    _Float16* wfrag = (_Float16*)d_ws;

    hivae_pack_w<<<(FRAG_HALF + 255) / 256, 256, 0, stream>>>(Wz, Wreal, Wcat, wfrag);
    // 2 waves/block, 16 rows/wave -> 32 rows/block
    hivae_main<<<BATCH / 32, 64, 0, stream>>>(y, s, soh, bz, brel, bcat, eps, gum,
                                              miss, wfrag, out);
}
